// Mata_6468220748539
// MI455X (gfx1250) — matrix operand order repaired and checked by reading, not compiled, not run on hardware
//
#include <hip/hip_runtime.h>
#include <hip/hip_bf16.h>
#include <math.h>

// ---------------- problem constants (match reference) ----------------
#define BQ      256
#define NN      128
#define NTOT    (BQ * NN)          // 32768
#define FD      128                // hidden dim
#define LDIM    29
#define MAXDEG  16
#define RWDIM   16
#define NLAB    (LDIM + MAXDEG + RWDIM)   // 61
#define SKL     (NN * NN)          // 16384 (Sinkhorn L)
#define LDSP    136                // LDS pitch in halves (272B, conflict-free-ish)
#define SKCHUNK 16128              // floats staged in LDS (63KB); tail stays in L2

typedef __attribute__((ext_vector_type(16))) _Float16 v16h;
typedef __attribute__((ext_vector_type(8)))  float    v8f;
typedef __attribute__((ext_vector_type(4)))  int      v4i;

// CDNA5 async global->LDS path (ASYNCcnt), guarded for toolchain portability.
// Builtin prototype (probed): (v4i addrspace(1)* src, v4i addrspace(3)* dst,
//                              imm int offset, imm int cpol)
#if __has_builtin(__builtin_amdgcn_global_load_async_to_lds_b128) && \
    __has_builtin(__builtin_amdgcn_s_wait_asynccnt)
#define HAVE_ASYNC_LDS 1
typedef __attribute__((address_space(1))) v4i* gv4i_p;
typedef __attribute__((address_space(3))) v4i* lv4i_p;
#else
#define HAVE_ASYNC_LDS 0
#endif

// =====================================================================
// WMMA GEMM:  H[M x 128] = (optional relu)(X[M x 128]) @ W[128 x 128]
// block = 256 threads (8 waves), covers 128 rows; W staged transposed
// in LDS as f16.  Each wave: 16 rows, 8 n-tiles x 4 k-steps = 32 WMMAs.
// =====================================================================
__global__ __launch_bounds__(256) void gemm_xw_kernel(
    const float* __restrict__ X, const float* __restrict__ W,
    float* __restrict__ H, int relu_in)
{
    __shared__ _Float16 Wt[FD * LDSP];
    for (int idx = threadIdx.x; idx < FD * FD; idx += 256) {
        int n = idx & 127, k = idx >> 7;
        Wt[n * LDSP + k] = (_Float16)W[k * FD + n];
    }
    __syncthreads();

    const int wave = threadIdx.x >> 5;
    const int lane = threadIdx.x & 31;
    const int m0   = blockIdx.x * 128 + wave * 16;

    // ---- load 4 A fragments (16x32 f16 each), held in VGPRs ----
    const int   arow = m0 + (lane & 15);
    const float* xr  = X + (size_t)arow * FD;
    const int   kb   = (lane >> 4) << 4;      // +16 for hi lane group
    v16h afr[4];
#pragma unroll
    for (int kt = 0; kt < 4; ++kt) {
        const float* p = xr + kt * 32 + kb;
        v16h a;
#pragma unroll
        for (int e = 0; e < 8; ++e) {
            float v0 = p[e], v1 = p[e + 8];
            if (relu_in) { v0 = fmaxf(v0, 0.f); v1 = fmaxf(v1, 0.f); }
            a[e] = (_Float16)v0; a[e + 8] = (_Float16)v1;
        }
        afr[kt] = a;
    }

    const int nloc  = lane & 15;
    const int khalf = (lane >> 4) << 4;       // contiguous 16-K run per lane grp
#pragma unroll
    for (int nt = 0; nt < 8; ++nt) {
        v8f acc = {};
#pragma unroll
        for (int kt = 0; kt < 4; ++kt) {
            const _Float16* bp = &Wt[(nt * 16 + nloc) * LDSP + kt * 32 + khalf];
            v16h bfr;
#pragma unroll
            for (int e = 0; e < 16; ++e) bfr[e] = bp[e];
            acc = __builtin_amdgcn_wmma_f32_16x16x32_f16(
                false, afr[kt], false, bfr, (short)0, acc, false, false);
        }
        const int col   = nt * 16 + nloc;
        const int rbase = m0 + ((lane >> 4) << 3);
#pragma unroll
        for (int r = 0; r < 8; ++r)
            H[(size_t)(rbase + r) * FD + col] = acc[r];
    }
}

// =====================================================================
// Batched affinity:  S[b] = Y_b (128x128) @ Xb_b^T   (one block / batch)
// B operand (Xb rows = columns of B) staged in LDS as f16.
// =====================================================================
__global__ __launch_bounds__(256) void bmm_affinity_kernel(
    const float* __restrict__ Y, const float* __restrict__ Xb,
    float* __restrict__ S)
{
    const int b = blockIdx.x;
    const float* Yb  = Y  + (size_t)b * NN * FD;
    const float* Xbb = Xb + (size_t)b * NN * FD;
    float*       Sb  = S  + (size_t)b * NN * NN;

    __shared__ _Float16 Bs[NN * LDSP];       // Bs[n][k] = Xb[b,n,k]
    for (int idx = threadIdx.x; idx < NN * FD; idx += 256) {
        int k = idx & 127, n = idx >> 7;
        Bs[n * LDSP + k] = (_Float16)Xbb[(size_t)n * FD + k];
    }
    __syncthreads();

    const int wave = threadIdx.x >> 5;
    const int lane = threadIdx.x & 31;
    const int m0   = wave * 16;

    const int   arow = m0 + (lane & 15);
    const float* yr  = Yb + (size_t)arow * FD;
    const int   kb   = (lane >> 4) << 4;
    v16h afr[4];
#pragma unroll
    for (int kt = 0; kt < 4; ++kt) {
        const float* p = yr + kt * 32 + kb;
        v16h a;
#pragma unroll
        for (int e = 0; e < 8; ++e) {
            a[e]     = (_Float16)p[e];
            a[e + 8] = (_Float16)p[e + 8];
        }
        afr[kt] = a;
    }

    const int nloc  = lane & 15;
    const int khalf = (lane >> 4) << 4;
#pragma unroll
    for (int nt = 0; nt < 8; ++nt) {
        v8f acc = {};
#pragma unroll
        for (int kt = 0; kt < 4; ++kt) {
            const _Float16* bp = &Bs[(nt * 16 + nloc) * LDSP + kt * 32 + khalf];
            v16h bfr;
#pragma unroll
            for (int e = 0; e < 16; ++e) bfr[e] = bp[e];
            acc = __builtin_amdgcn_wmma_f32_16x16x32_f16(
                false, afr[kt], false, bfr, (short)0, acc, false, false);
        }
        const int col   = nt * 16 + nloc;
        const int rbase = m0 + ((lane >> 4) << 3);
#pragma unroll
        for (int r = 0; r < 8; ++r)
            Sb[(size_t)(rbase + r) * NN + col] = acc[r];
    }
}

// =====================================================================
// Init embedding: feat = relu(concat[x, demb[cent], rw] @ initW + b)
// =====================================================================
__global__ __launch_bounds__(256) void init_feat_kernel(
    const float* __restrict__ x, const int* __restrict__ cent,
    const float* __restrict__ rw, const float* __restrict__ demb,
    const float* __restrict__ W, const float* __restrict__ bias,
    float* __restrict__ out)
{
    int idx = blockIdx.x * 256 + threadIdx.x;       // NTOT*FD threads
    int n = idx >> 7, j = idx & 127;
    float acc = bias[j];
    const float* xr = x + (size_t)n * LDIM;
#pragma unroll 4
    for (int k = 0; k < LDIM; ++k)  acc += xr[k] * W[k * FD + j];
    const float* dr = demb + (size_t)cent[n] * MAXDEG;
#pragma unroll 4
    for (int k = 0; k < MAXDEG; ++k) acc += dr[k] * W[(LDIM + k) * FD + j];
    const float* rr = rw + (size_t)n * RWDIM;
#pragma unroll 4
    for (int k = 0; k < RWDIM; ++k) acc += rr[k] * W[(LDIM + MAXDEG + k) * FD + j];
    out[idx] = fmaxf(acc, 0.f);
}

// ---------------- degree handling ----------------
__global__ void fill_kernel(float* __restrict__ p, float v, int n)
{
    int i = blockIdx.x * 256 + threadIdx.x;
    if (i < n) p[i] = v;
}
__global__ void deg_edge_kernel(const int* __restrict__ dst,
                                float* __restrict__ deg, int E)
{
    int e = blockIdx.x * 256 + threadIdx.x;
    if (e < E) atomicAdd(&deg[dst[e]], 1.0f);
}
__global__ void dinv_kernel(float* __restrict__ deg, int n)
{
    int i = blockIdx.x * 256 + threadIdx.x;
    if (i < n) deg[i] = rsqrtf(fmaxf(deg[i], 1.0f));
}

// ---------------- GCN scatter: self-loop init then edge atomics ----------------
__global__ __launch_bounds__(256) void conv_self_kernel(
    const float* __restrict__ h, const float* __restrict__ dinv,
    const float* __restrict__ bias, float* __restrict__ out)
{
    int idx = blockIdx.x * 256 + threadIdx.x;       // NTOT*FD
    int n = idx >> 7, f = idx & 127;
    float w = dinv[n]; w *= w;
    out[idx] = h[idx] * w + bias[f];
}
__global__ __launch_bounds__(128) void conv_edge_kernel(
    const float* __restrict__ h, const int* __restrict__ src,
    const int* __restrict__ dst, const float* __restrict__ dinv,
    float* __restrict__ out)
{
    int e = blockIdx.x;
    int f = threadIdx.x;
    int s = src[e], d = dst[e];
    float w = dinv[s] * dinv[d];
    atomicAdd(&out[(size_t)d * FD + f], h[(size_t)s * FD + f] * w);
}

// =====================================================================
// Sinkhorn soft-topk, 2 anchors.  f collapses to an elementwise function
// of (x_i, g) so state is just the 2-vector g -> pure block reductions.
// One block (256 thr) per batch.  The 64KB score tile is staged into LDS
// once via CDNA5 async global->LDS (ASYNCcnt) -- it is re-read ~8x, so
// staging converts 8 L2 round-trips/element into 1.  Scores updated in
// place (each thread only touches its own strided elements).
// =====================================================================
__global__ __launch_bounds__(256) void sinkhorn_kernel(
    float* __restrict__ S, const int* __restrict__ topk)
{
    const int L = SKL;
    float* x = S + (size_t)blockIdx.x * L;
    const int tid = threadIdx.x, lane = tid & 31, wv = tid >> 5;
    __shared__ __attribute__((aligned(16))) float xs[SKCHUNK];
    __shared__ float sA[8], sB[8], sC[8], sD[8];
    __shared__ float bc[2];

    // ---- stage first 63KB of the tile into LDS; 1KB tail stays in L2 ----
#if HAVE_ASYNC_LDS
    for (int c = tid; c < SKCHUNK / 4; c += 256) {
        __builtin_amdgcn_global_load_async_to_lds_b128(
            (gv4i_p)(x + 4 * c), (lv4i_p)(xs + 4 * c), 0, 0);
    }
    __builtin_amdgcn_s_wait_asynccnt(0);
#else
    for (int c = tid; c < SKCHUNK; c += 256) xs[c] = x[c];
#endif
    __syncthreads();

    // anchors from min/max
    float mn = 3.0e38f, mx = -3.0e38f;
    for (int i = tid; i < L; i += 256) {
        float v = (i < SKCHUNK) ? xs[i] : x[i];
        mn = fminf(mn, v); mx = fmaxf(mx, v);
    }
    for (int off = 16; off; off >>= 1) {
        mn = fminf(mn, __shfl_xor(mn, off, 32));
        mx = fmaxf(mx, __shfl_xor(mx, off, 32));
    }
    if (lane == 0) { sA[wv] = mn; sB[wv] = mx; }
    __syncthreads();
    if (tid == 0) {
        float a = sA[0], m = sB[0];
        for (int k = 1; k < 8; ++k) { a = fminf(a, sA[k]); m = fmaxf(m, sB[k]); }
        bc[0] = a - 1.0f; bc[1] = m + 1.0f;
    }
    __syncthreads();
    const float a0 = bc[0], a1 = bc[1];            // SK_TAU == 1.0

    const float kk      = (float)topk[0] * 0.5f;
    const float log_nu0 = logf(((float)L - kk) / (float)L);
    const float log_nu1 = logf(kk / (float)L);
    const float log_mu  = -logf((float)L);

    float g0 = 0.f, g1 = 0.f, gp0 = 0.f, gp1 = 0.f;
    for (int it = 0; it < 6; ++it) {
        gp0 = g0; gp1 = g1;
        float m0 = -3.0e38f, s0 = 0.f, m1 = -3.0e38f, s1 = 0.f;
        for (int i = tid; i < L; i += 256) {
            float v  = (i < SKCHUNK) ? xs[i] : x[i];
            float k0 = -fabsf(v - a0), k1 = -fabsf(v - a1);
            float t0 = k0 + gp0, t1 = k1 + gp1;
            float M  = fmaxf(t0, t1);
            float f  = log_mu - (M + logf(expf(t0 - M) + expf(t1 - M)));
            float u0 = k0 + f;
            if (u0 > m0) { s0 = s0 * expf(m0 - u0) + 1.f; m0 = u0; }
            else         { s0 += expf(u0 - m0); }
            float u1 = k1 + f;
            if (u1 > m1) { s1 = s1 * expf(m1 - u1) + 1.f; m1 = u1; }
            else         { s1 += expf(u1 - m1); }
        }
        for (int off = 16; off; off >>= 1) {
            float m2 = __shfl_xor(m0, off, 32), s2 = __shfl_xor(s0, off, 32);
            float M  = fmaxf(m0, m2); s0 = s0 * expf(m0 - M) + s2 * expf(m2 - M); m0 = M;
            m2 = __shfl_xor(m1, off, 32); s2 = __shfl_xor(s1, off, 32);
            M  = fmaxf(m1, m2); s1 = s1 * expf(m1 - M) + s2 * expf(m2 - M); m1 = M;
        }
        __syncthreads();
        if (lane == 0) { sA[wv] = m0; sB[wv] = s0; sC[wv] = m1; sD[wv] = s1; }
        __syncthreads();
        if (tid == 0) {
            float M0 = sA[0], S0 = sB[0], M1 = sC[0], S1 = sD[0];
            for (int k = 1; k < 8; ++k) {
                float M = fmaxf(M0, sA[k]);
                S0 = S0 * expf(M0 - M) + sB[k] * expf(sA[k] - M); M0 = M;
                M  = fmaxf(M1, sC[k]);
                S1 = S1 * expf(M1 - M) + sD[k] * expf(sC[k] - M); M1 = M;
            }
            bc[0] = M0 + logf(S0); bc[1] = M1 + logf(S1);
        }
        __syncthreads();
        g0 = log_nu0 - bc[0];
        g1 = log_nu1 - bc[1];
        __syncthreads();
    }
    // prob = exp(logK[:,1] + f(gp) + g1) * L  (f uses second-to-last g)
    for (int i = tid; i < L; i += 256) {
        float v  = (i < SKCHUNK) ? xs[i] : x[i];
        float k0 = -fabsf(v - a0), k1 = -fabsf(v - a1);
        float t0 = k0 + gp0, t1 = k1 + gp1;
        float M  = fmaxf(t0, t1);
        float f  = log_mu - (M + logf(expf(t0 - M) + expf(t1 - M)));
        x[i] = expf(k1 + f + g1) * (float)L;
    }
}

// ---------------- pooled global features + scoring MLP ----------------
__global__ __launch_bounds__(256) void global_feat_kernel(
    const float* __restrict__ f10, const float* __restrict__ f11,
    const float* __restrict__ f12, const float* __restrict__ f13,
    const float* __restrict__ f20, const float* __restrict__ f21,
    const float* __restrict__ f22, const float* __restrict__ f23,
    float* __restrict__ gf)
{
    int idx = blockIdx.x * 256 + threadIdx.x;       // BQ*1024
    int b = idx >> 10, c = idx & 1023;
    const float* m1[4] = { f10, f11, f12, f13 };
    const float* m2[4] = { f20, f21, f22, f23 };
    float r;
    if (c < 512) {                                  // sum-pool graph 1
        const float* m = m1[c >> 7];
        int col = c & 127;
        r = 0.f;
        for (int n = 0; n < NN; ++n)
            r += m[((size_t)b * NN + n) * FD + col];
    } else {                                        // max-pool graph 2
        const float* m = m2[(c - 512) >> 7];
        int col = (c - 512) & 127;
        r = -3.0e38f;
        for (int n = 0; n < NN; ++n)
            r = fmaxf(r, m[((size_t)b * NN + n) * FD + col]);
    }
    gf[idx] = r;
}
__global__ __launch_bounds__(64) void mlp1_kernel(
    const float* __restrict__ gf, const float* __restrict__ W,
    const float* __restrict__ b, float* __restrict__ h)
{
    int idx = blockIdx.x * 64 + threadIdx.x;        // BQ*64
    int row = idx >> 6, j = idx & 63;
    float acc = b[j];
    const float* g = gf + (size_t)row * 1024;
#pragma unroll 4
    for (int c = 0; c < 1024; ++c) acc += g[c] * W[c * 64 + j];
    h[idx] = fmaxf(acc, 0.f);
}
__global__ __launch_bounds__(64) void mlp2_kernel(
    const float* __restrict__ h, const float* __restrict__ W,
    const float* __restrict__ b, float* __restrict__ ged)
{
    int r = blockIdx.x * 64 + threadIdx.x;
    if (r < BQ) {
        float acc = b[0];
#pragma unroll 8
        for (int j = 0; j < 64; ++j) acc += h[(size_t)r * 64 + j] * W[j];
        ged[r] = 1.f / (1.f + expf(-acc));
    }
}

// =====================================================================
extern "C" void kernel_launch(void* const* d_in, const int* in_sizes, int n_in,
                              void* d_out, int out_size, void* d_ws, size_t ws_size,
                              hipStream_t stream)
{
    const float* x1    = (const float*)d_in[0];
    const int*   cent1 = (const int*)  d_in[1];
    const float* rw1   = (const float*)d_in[2];
    const int*   src1  = (const int*)  d_in[3];
    const int*   dst1  = (const int*)  d_in[4];
    const float* x2    = (const float*)d_in[5];
    const int*   cent2 = (const int*)  d_in[6];
    const float* rw2   = (const float*)d_in[7];
    const int*   src2  = (const int*)  d_in[8];
    const int*   dst2  = (const int*)  d_in[9];
    const float* demb  = (const float*)d_in[10];
    const float* initW = (const float*)d_in[11];
    const float* initb = (const float*)d_in[12];
    const float* W1    = (const float*)d_in[13];
    const float* b1    = (const float*)d_in[14];
    const float* W2    = (const float*)d_in[15];
    const float* b2    = (const float*)d_in[16];
    const float* W3    = (const float*)d_in[17];
    const float* b3    = (const float*)d_in[18];
    const float* Aaff  = (const float*)d_in[19];
    const float* scW1  = (const float*)d_in[20];
    const float* scb1  = (const float*)d_in[21];
    const float* scW2  = (const float*)d_in[22];
    const float* scb2  = (const float*)d_in[23];
    const int*   topk  = (const int*)  d_in[24];
    const int E = in_sizes[3];

    // ---- workspace carve ----
    const size_t NF = (size_t)NTOT * FD;   // 4,194,304 floats
    float* ws = (float*)d_ws;
    float* feat1 = ws; ws += NF;
    float* feat2 = ws; ws += NF;
    float* f11   = ws; ws += NF;
    float* f12   = ws; ws += NF;
    float* f13   = ws; ws += NF;
    float* f21   = ws; ws += NF;
    float* f22   = ws; ws += NF;
    float* f23   = ws; ws += NF;
    float* htmp  = ws; ws += NF;           // GEMM output / affinity Y
    float* deg1  = ws; ws += NTOT;
    float* deg2  = ws; ws += NTOT;
    float* gf    = ws; ws += (size_t)BQ * 1024;
    float* hbuf  = ws; ws += (size_t)BQ * 64;

    float* out  = (float*)d_out;
    float* ged  = out;
    float* sim1 = out + BQ;
    float* sim2 = sim1 + (size_t)BQ * NN * NN;

    const int nodeThreads = NTOT * FD;     // 4,194,304

    // ---- init embeddings ----
    init_feat_kernel<<<nodeThreads / 256, 256, 0, stream>>>(x1, cent1, rw1, demb, initW, initb, feat1);
    init_feat_kernel<<<nodeThreads / 256, 256, 0, stream>>>(x2, cent2, rw2, demb, initW, initb, feat2);

    // ---- degrees (self-loop counts as 1) -> dinv in place ----
    fill_kernel<<<(NTOT + 255) / 256, 256, 0, stream>>>(deg1, 1.0f, NTOT);
    fill_kernel<<<(NTOT + 255) / 256, 256, 0, stream>>>(deg2, 1.0f, NTOT);
    deg_edge_kernel<<<(E + 255) / 256, 256, 0, stream>>>(dst1, deg1, E);
    deg_edge_kernel<<<(E + 255) / 256, 256, 0, stream>>>(dst2, deg2, E);
    dinv_kernel<<<(NTOT + 255) / 256, 256, 0, stream>>>(deg1, NTOT);
    dinv_kernel<<<(NTOT + 255) / 256, 256, 0, stream>>>(deg2, NTOT);

    // ---- GCN conv layer helper ----
    auto conv = [&](const float* X, int relu_in, const float* W, const float* bb,
                    const int* src, const int* dst, const float* dinv, float* outF) {
        gemm_xw_kernel<<<NTOT / 128, 256, 0, stream>>>(X, W, htmp, relu_in);
        conv_self_kernel<<<nodeThreads / 256, 256, 0, stream>>>(htmp, dinv, bb, outF);
        conv_edge_kernel<<<E, 128, 0, stream>>>(htmp, src, dst, dinv, outF);
    };

    conv(feat1, 0, W1, b1, src1, dst1, deg1, f11);
    conv(f11,   1, W2, b2, src1, dst1, deg1, f12);
    conv(f12,   1, W3, b3, src1, dst1, deg1, f13);
    conv(feat2, 0, W1, b1, src2, dst2, deg2, f21);
    conv(f21,   1, W2, b2, src2, dst2, deg2, f22);
    conv(f22,   1, W3, b3, src2, dst2, deg2, f23);

    // ---- affinity + Sinkhorn #1 (feat1 vs feat2) ----
    gemm_xw_kernel<<<NTOT / 128, 256, 0, stream>>>(feat1, Aaff, htmp, 0);
    bmm_affinity_kernel<<<BQ, 256, 0, stream>>>(htmp, feat2, sim1);
    sinkhorn_kernel<<<BQ, 256, 0, stream>>>(sim1, topk);

    // ---- affinity + Sinkhorn #2 (f13 vs f23) ----
    gemm_xw_kernel<<<NTOT / 128, 256, 0, stream>>>(f13, Aaff, htmp, 0);
    bmm_affinity_kernel<<<BQ, 256, 0, stream>>>(htmp, f23, sim2);
    sinkhorn_kernel<<<BQ, 256, 0, stream>>>(sim2, topk);

    // ---- pooled features + scoring MLP ----
    global_feat_kernel<<<(BQ * 1024) / 256, 256, 0, stream>>>(
        feat1, f11, f12, f13, feat2, f21, f22, f23, gf);
    mlp1_kernel<<<BQ, 64, 0, stream>>>(gf, scW1, scb1, hbuf);
    mlp2_kernel<<<(BQ + 63) / 64, 64, 0, stream>>>(hbuf, scW2, scb2, ged);
}
